// Encoderlayer_60000693125600
// MI455X (gfx1250) — matrix operand order repaired and checked by reading, not compiled, not run on hardware
//
#include <hip/hip_runtime.h>

// ---------------------------------------------------------------------------
// MTGNN-style encoder layer for MI455X (gfx1250), wave32 + WMMA bf16.
// Heavy graph-propagation GEMMs (A[1000x1000] x H[1000 x B*C*t]) run on
// v_wmma_f32_16x16x32_bf16, register-blocked 2x4 tiles per wave (24 B of
// fragment traffic per WMMA), fused mixprop epilogue. Compile-only target;
// fragment layouts follow CDNA5 ISA.
// ---------------------------------------------------------------------------

typedef __attribute__((ext_vector_type(16))) __bf16 v16bf;
typedef __attribute__((ext_vector_type(8)))  __bf16 v8bf;
typedef __attribute__((ext_vector_type(8)))  float  v8f;

static constexpr int   B_   = 16;
static constexpr int   C_   = 32;
static constexpr int   N_   = 1000;
static constexpr int   T_   = 72;
static constexpr int   M_   = 200;
static constexpr int   ND_  = 40;
static constexpr int   NP_  = 1024;   // padded node count (K and M for mixprop GEMM)
static constexpr int   MPA_ = 224;    // padded 200 for assign GEMM (multiple of 32)
static constexpr float TANHA = 3.0f;
static constexpr float PALPHA = 0.05f;
static constexpr float EPSLN = 1e-5f;

// ------------------------------ small kernels ------------------------------

__global__ void nv_kernel(const float* __restrict__ emb, const float* __restrict__ W,
                          const float* __restrict__ bias, float* __restrict__ nv) {
  int i = blockIdx.x * blockDim.x + threadIdx.x;
  if (i >= N_ * ND_) return;
  int n = i / ND_, j = i % ND_;
  const float* e = emb + n * ND_;
  const float* w = W + j * ND_;          // (emb @ W^T)[n,j]
  float s = bias[j];
  for (int d = 0; d < ND_; ++d) s += e[d] * w[d];
  nv[i] = tanhf(TANHA * s);
}

__global__ void adj_kernel(const float* __restrict__ nv1, const float* __restrict__ nv2,
                           float* __restrict__ adp) {
  int i = blockIdx.x * blockDim.x + threadIdx.x;
  if (i >= N_ * N_) return;
  int v = i / N_, w = i % N_;
  const float* a1 = nv1 + v * ND_;
  const float* a2 = nv2 + v * ND_;
  const float* b1 = nv1 + w * ND_;
  const float* b2 = nv2 + w * ND_;
  float s = 0.f;
  for (int d = 0; d < ND_; ++d) s += a1[d] * b2[d] - a2[d] * b1[d];
  adp[i] = fmaxf(0.f, tanhf(TANHA * s));
}

// one block per row: keep top-20 entries, zero the rest (in place)
__global__ void __launch_bounds__(128) topk_kernel(float* __restrict__ adp) {
  __shared__ float vals[N_];
  __shared__ float rmax[128];
  __shared__ int   ridx[128];
  int v = blockIdx.x, tid = threadIdx.x;
  float* row = adp + (size_t)v * N_;
  for (int i = tid; i < N_; i += 128) vals[i] = row[i];
  __syncthreads();
  for (int it = 0; it < 20; ++it) {
    float m = -1.f; int mi = -1;
    for (int i = tid; i < N_; i += 128)
      if (vals[i] > m) { m = vals[i]; mi = i; }
    rmax[tid] = m; ridx[tid] = mi;
    __syncthreads();
    for (int s = 64; s > 0; s >>= 1) {
      if (tid < s && rmax[tid + s] > rmax[tid]) { rmax[tid] = rmax[tid + s]; ridx[tid] = ridx[tid + s]; }
      __syncthreads();
    }
    if (tid == 0 && ridx[0] >= 0) vals[ridx[0]] = -2.f;   // mark selected (adj >= 0)
    __syncthreads();
  }
  for (int i = tid; i < N_; i += 128)
    if (vals[i] > -1.5f) row[i] = 0.f;                    // not selected -> zero
}

// rs1[v] = 1 + sum_w adp[v,w]   (row sums, for A1)
// rs2[v] = 1 + sum_w adp[w,v]   (col sums, for A2 = adp^T normalization)
__global__ void __launch_bounds__(128) rowsum_kernel(const float* __restrict__ adp,
                                                     float* __restrict__ rs1,
                                                     float* __restrict__ rs2) {
  __shared__ float s1[128], s2[128];
  int v = blockIdx.x, tid = threadIdx.x;
  float a = 0.f, b = 0.f;
  for (int w = tid; w < N_; w += 128) {
    a += adp[(size_t)v * N_ + w];
    b += adp[(size_t)w * N_ + v];
  }
  s1[tid] = a; s2[tid] = b;
  __syncthreads();
  for (int s = 64; s > 0; s >>= 1) {
    if (tid < s) { s1[tid] += s1[tid + s]; s2[tid] += s2[tid + s]; }
    __syncthreads();
  }
  if (tid == 0) { rs1[v] = s1[0] + 1.f; rs2[v] = s2[0] + 1.f; }
}

// A1[v,w] = (adp[v,w] + I)/rs1[v],  A2[v,w] = (adp[w,v] + I)/rs2[v], bf16, zero-padded
__global__ void adjpack_kernel(const float* __restrict__ adp, const float* __restrict__ rs1,
                               const float* __restrict__ rs2,
                               __bf16* __restrict__ A1, __bf16* __restrict__ A2) {
  int i = blockIdx.x * blockDim.x + threadIdx.x;
  if (i >= NP_ * NP_) return;
  int v = i / NP_, w = i % NP_;
  float a1 = 0.f, a2 = 0.f;
  if (v < N_ && w < N_) {
    float eye = (v == w) ? 1.f : 0.f;
    a1 = (adp[(size_t)v * N_ + w] + eye) / rs1[v];
    a2 = (adp[(size_t)w * N_ + v] + eye) / rs2[v];
  }
  A1[i] = (__bf16)a1;
  A2[i] = (__bf16)a2;
}

// ------------------------- gated dilated inception -------------------------

__device__ __forceinline__ void lds_copy(float* dst, const float* src, int n, int tid, int nt) {
  for (int i = tid; i < n; i += nt) dst[i] = src[i];
}

__global__ void __launch_bounds__(256)
inception_kernel(const float* __restrict__ xin, int Tin, int tout,
                 const float* fw0, const float* fw1, const float* fw2, const float* fw3,
                 const float* fb0, const float* fb1, const float* fb2, const float* fb3,
                 const float* gw0, const float* gw1, const float* gw2, const float* gw3,
                 const float* gb0, const float* gb1, const float* gb2, const float* gb3,
                 float* __restrict__ xg) {
  // branches: k = 2,3,7,13 ; cout = 8 each ; weights [co][ci][tau]
  __shared__ float s_fw[6400];
  __shared__ float s_gw[6400];
  __shared__ float s_fb[32];
  __shared__ float s_gb[32];
  const int tid = threadIdx.x, nt = blockDim.x;
  lds_copy(s_fw + 0,    fw0, 8 * C_ * 2,  tid, nt);
  lds_copy(s_fw + 512,  fw1, 8 * C_ * 3,  tid, nt);
  lds_copy(s_fw + 1280, fw2, 8 * C_ * 7,  tid, nt);
  lds_copy(s_fw + 3072, fw3, 8 * C_ * 13, tid, nt);
  lds_copy(s_gw + 0,    gw0, 8 * C_ * 2,  tid, nt);
  lds_copy(s_gw + 512,  gw1, 8 * C_ * 3,  tid, nt);
  lds_copy(s_gw + 1280, gw2, 8 * C_ * 7,  tid, nt);
  lds_copy(s_gw + 3072, gw3, 8 * C_ * 13, tid, nt);
  lds_copy(s_fb + 0, fb0, 8, tid, nt); lds_copy(s_fb + 8,  fb1, 8, tid, nt);
  lds_copy(s_fb + 16, fb2, 8, tid, nt); lds_copy(s_fb + 24, fb3, 8, tid, nt);
  lds_copy(s_gb + 0, gb0, 8, tid, nt); lds_copy(s_gb + 8,  gb1, 8, tid, nt);
  lds_copy(s_gb + 16, gb2, 8, tid, nt); lds_copy(s_gb + 24, gb3, 8, tid, nt);
  __syncthreads();

  int gid = blockIdx.x * blockDim.x + tid;
  int total = B_ * N_ * tout;
  if (gid >= total) return;
  int j = gid % tout;
  int v = (gid / tout) % N_;
  int b = gid / (tout * N_);

  const int KK[4]   = {2, 3, 7, 13};
  const int BOFF[4] = {0, 512, 1280, 3072};

  float fa[32], ga[32];
  #pragma unroll
  for (int o = 0; o < 32; ++o) { fa[o] = 0.f; ga[o] = 0.f; }

  for (int ci = 0; ci < C_; ++ci) {
    const float* xp = xin + ((size_t)(b * C_ + ci) * N_ + v) * Tin + j;  // window [j, j+12]
    float w13[13];
    #pragma unroll
    for (int t = 0; t < 13; ++t) w13[t] = xp[t];
    #pragma unroll
    for (int bi = 0; bi < 4; ++bi) {
      const int k = KK[bi];
      #pragma unroll
      for (int co = 0; co < 8; ++co) {
        const float* wf = &s_fw[BOFF[bi] + (co * C_ + ci) * k];
        const float* wg = &s_gw[BOFF[bi] + (co * C_ + ci) * k];
        float sf = 0.f, sg = 0.f;
        #pragma unroll
        for (int tau = 0; tau < k; ++tau) {
          float xv = w13[13 - k + tau];
          sf += wf[tau] * xv; sg += wg[tau] * xv;
        }
        fa[bi * 8 + co] += sf; ga[bi * 8 + co] += sg;
      }
    }
  }
  #pragma unroll
  for (int oc = 0; oc < 32; ++oc) {
    float f = tanhf(fa[oc] + s_fb[oc]);
    float g = 1.f / (1.f + expf(-(ga[oc] + s_gb[oc])));
    xg[((size_t)(b * C_ + oc) * N_ + v) * tout + j] = f * g;
  }
}

// ----------------------------- packing kernels -----------------------------

// src [B,C,N,tlen] f32 -> dst [L, NP] bf16 (row l=(b*C+c)*tlen+t, col v; v>=N zero)
__global__ void pack_h_kernel(const float* __restrict__ src, __bf16* __restrict__ dst,
                              int tlen, int L) {
  size_t i = (size_t)blockIdx.x * blockDim.x + threadIdx.x;
  if (i >= (size_t)L * NP_) return;
  int v = (int)(i & (NP_ - 1));
  int l = (int)(i >> 10);
  int bc = l / tlen, t = l % tlen;
  float val = (v < N_) ? src[((size_t)bc * N_ + v) * tlen + t] : 0.f;
  dst[i] = (__bf16)val;
}

// assign [N,M] f32 -> AssT [MPA, NP] bf16 (row m, col n)
__global__ void pack_assT_kernel(const float* __restrict__ ass, __bf16* __restrict__ dst) {
  int i = blockIdx.x * blockDim.x + threadIdx.x;
  if (i >= MPA_ * NP_) return;
  int m = i / NP_, n = i % NP_;
  float v = (m < M_ && n < N_) ? ass[(size_t)n * M_ + m] : 0.f;
  dst[i] = (__bf16)v;
}

// ------------------------------- WMMA GEMM ---------------------------------
// D = A x B,  A [Mp x Kp] row-major bf16, B given transposed: Bt [L x Kp].
// Each wave computes a 2x4 block of 16x16 tiles (32 rows x 64 cols):
//   per k-step: 2 A-frag loads (64B) + 4 B-frag loads (128B) -> 8 WMMAs
//   (24 B of operand traffic per WMMA; 8 independent accumulators).
// Output stored column-tile friendly: outF[l*Mp + m] (f32).
// If X != null (mix mode): out = PALPHA * X[l,m] + (1-PALPHA) * D  (X bf16 [L x Mp]).
// If outB != null: also store bf16 of result (for next hop's B operand).
// Requires Mp % 32 == 0, L % 64 == 0, Kp % 32 == 0.
__global__ void __launch_bounds__(32)
gemm_bf16_wmma(const __bf16* __restrict__ A, const __bf16* __restrict__ Bt,
               const __bf16* __restrict__ X, float* __restrict__ outF,
               __bf16* __restrict__ outB, int Mp, int Kp, int L, int mblocks) {
  int tile = blockIdx.x;
  int tm = tile % mblocks;              // 32-row block index
  int tl = tile / mblocks;              // 64-col block index
  int lane = threadIdx.x;
  int r16  = lane & 15;
  int hi   = (lane < 16) ? 0 : 1;

  // A fragment rows: lane m = base + r16; k runs {hi*8..hi*8+7} and {16+hi*8..}
  const __bf16* arow[2];
  arow[0] = A + (size_t)(tm * 32 + r16) * Kp + hi * 8;
  arow[1] = arow[0] + (size_t)16 * Kp;
  // B fragment rows: lane n = base + r16; k runs {hi*8..hi*8+7} and {16+hi*8..}
  const __bf16* brow[4];
  brow[0] = Bt + (size_t)(tl * 64 + r16) * Kp + hi * 8;
  brow[1] = brow[0] + (size_t)16 * Kp;
  brow[2] = brow[0] + (size_t)32 * Kp;
  brow[3] = brow[0] + (size_t)48 * Kp;

  v8f acc[2][4];
  #pragma unroll
  for (int im = 0; im < 2; ++im)
    #pragma unroll
    for (int il = 0; il < 4; ++il) acc[im][il] = (v8f){};

  for (int kb = 0; kb < Kp; kb += 32) {
    v16bf av[2], bv[4];
    #pragma unroll
    for (int im = 0; im < 2; ++im) {
      v8bf a0 = *(const v8bf*)(arow[im] + kb);
      v8bf a1 = *(const v8bf*)(arow[im] + kb + 16);
      #pragma unroll
      for (int i = 0; i < 8; ++i) { av[im][i] = a0[i]; av[im][8 + i] = a1[i]; }
    }
    #pragma unroll
    for (int il = 0; il < 4; ++il) {
      v8bf b0 = *(const v8bf*)(brow[il] + kb);
      v8bf b1 = *(const v8bf*)(brow[il] + kb + 16);
      #pragma unroll
      for (int i = 0; i < 8; ++i) { bv[il][i] = b0[i]; bv[il][8 + i] = b1[i]; }
    }
    #pragma unroll
    for (int im = 0; im < 2; ++im)
      #pragma unroll
      for (int il = 0; il < 4; ++il)
        acc[im][il] = __builtin_amdgcn_wmma_f32_16x16x32_bf16(
            /*neg_a=*/false, av[im], /*neg_b=*/false, bv[il],
            /*c_mod=*/(short)0, acc[im][il], /*reuse_a=*/false, /*reuse_b=*/false);
  }

  // D layout per tile: vgpr r -> m = mtile*16 + hi*8 + r ; col l = ltile*16 + r16
  #pragma unroll
  for (int im = 0; im < 2; ++im) {
    size_t mbase = (size_t)tm * 32 + im * 16 + hi * 8;
    #pragma unroll
    for (int il = 0; il < 4; ++il) {
      size_t l = (size_t)tl * 64 + il * 16 + r16;
      v8f d = acc[im][il];
      if (X) {
        v8bf xv = *(const v8bf*)(X + l * Mp + mbase);
        #pragma unroll
        for (int r = 0; r < 8; ++r) d[r] = PALPHA * (float)xv[r] + (1.f - PALPHA) * d[r];
      }
      *(v8f*)(outF + l * Mp + mbase) = d;
      if (outB) {
        v8bf ob;
        #pragma unroll
        for (int r = 0; r < 8; ++r) ob[r] = (__bf16)d[r];
        *(v8bf*)(outB + l * Mp + mbase) = ob;
      }
    }
  }
}

// ------------------------- mixprop projection accum -------------------------

// In place over xg: xg <- bias1+bias2 + residual + (W1+W2)[:, 0:C] * xg   (hop 0)
__global__ void __launch_bounds__(256)
acc_init_kernel(float* __restrict__ xg, const float* __restrict__ xres, int Tin, int tlen,
                const float* __restrict__ W1, const float* __restrict__ b1v,
                const float* __restrict__ W2, const float* __restrict__ b2v) {
  __shared__ float Ws[C_ * C_];
  __shared__ float bs[C_];
  for (int i = threadIdx.x; i < C_ * C_; i += blockDim.x) {
    int o = i / C_, c = i % C_;
    Ws[i] = W1[o * (3 * C_) + c] + W2[o * (3 * C_) + c];
  }
  for (int i = threadIdx.x; i < C_; i += blockDim.x) bs[i] = b1v[i] + b2v[i];
  __syncthreads();
  int gid = blockIdx.x * blockDim.x + threadIdx.x;
  int total = B_ * N_ * tlen;
  if (gid >= total) return;
  int t = gid % tlen;
  int v = (gid / tlen) % N_;
  int b = gid / (tlen * N_);
  int toff = Tin - tlen;
  float xv[C_];
  #pragma unroll
  for (int c = 0; c < C_; ++c) xv[c] = xg[((size_t)(b * C_ + c) * N_ + v) * tlen + t];
  float out[C_];
  #pragma unroll
  for (int o = 0; o < C_; ++o) {
    float s = bs[o] + xres[((size_t)(b * C_ + o) * N_ + v) * Tin + toff + t];
    #pragma unroll
    for (int c = 0; c < C_; ++c) s += Ws[o * C_ + c] * xv[c];
    out[o] = s;
  }
  #pragma unroll
  for (int o = 0; o < C_; ++o) xg[((size_t)(b * C_ + o) * N_ + v) * tlen + t] = out[o];
}

// acc += W[:, coff:coff+C] * h,  h read from Dt [L x NP] (f32 hop, transposed layout)
__global__ void __launch_bounds__(256)
acc_hop_kernel(float* __restrict__ acc, const float* __restrict__ Dt,
               const float* __restrict__ W, int coff, int tlen) {
  __shared__ float Ws[C_ * C_];
  for (int i = threadIdx.x; i < C_ * C_; i += blockDim.x) {
    int o = i / C_, c = i % C_;
    Ws[i] = W[o * (3 * C_) + coff + c];
  }
  __syncthreads();
  int gid = blockIdx.x * blockDim.x + threadIdx.x;
  int total = B_ * N_ * tlen;
  if (gid >= total) return;
  int t = gid % tlen;
  int v = (gid / tlen) % N_;
  int b = gid / (tlen * N_);
  float hv[C_];
  #pragma unroll
  for (int c = 0; c < C_; ++c)
    hv[c] = Dt[(size_t)((b * C_ + c) * tlen + t) * NP_ + v];
  #pragma unroll
  for (int o = 0; o < C_; ++o) {
    float s = 0.f;
    #pragma unroll
    for (int c = 0; c < C_; ++c) s += Ws[o * C_ + c] * hv[c];
    acc[((size_t)(b * C_ + o) * N_ + v) * tlen + t] += s;
  }
}

// -------------------------------- layernorm --------------------------------

__global__ void stats_zero_kernel(float* stats) {
  if (threadIdx.x < 32) stats[threadIdx.x] = 0.f;
}

__global__ void __launch_bounds__(256)
ln_reduce_kernel(const float* __restrict__ x, float* __restrict__ stats, int cnt) {
  __shared__ float ss[256], sq[256];
  int b = blockIdx.y;
  const float* xb = x + (size_t)b * cnt;
  float s = 0.f, q = 0.f;
  for (int i = blockIdx.x * blockDim.x + threadIdx.x; i < cnt; i += gridDim.x * blockDim.x) {
    float v = xb[i];
    s += v; q += v * v;
  }
  ss[threadIdx.x] = s; sq[threadIdx.x] = q;
  __syncthreads();
  for (int st = 128; st > 0; st >>= 1) {
    if (threadIdx.x < st) { ss[threadIdx.x] += ss[threadIdx.x + st]; sq[threadIdx.x] += sq[threadIdx.x + st]; }
    __syncthreads();
  }
  if (threadIdx.x == 0) {
    atomicAdd(&stats[b], ss[0]);
    atomicAdd(&stats[16 + b], sq[0]);
  }
}

__global__ void ln_apply_kernel(const float* __restrict__ x, const float* __restrict__ stats,
                                const float* __restrict__ lnw, const float* __restrict__ lnb,
                                float* __restrict__ out, int cnt) {
  size_t i = (size_t)blockIdx.x * blockDim.x + threadIdx.x;
  if (i >= (size_t)B_ * cnt) return;
  int b = (int)(i / cnt);
  int r = (int)(i % cnt);
  float inv = 1.f / (float)cnt;
  float mu  = stats[b] * inv;
  float var = stats[16 + b] * inv - mu * mu;
  out[i] = (x[i] - mu) * rsqrtf(var + EPSLN) * lnw[r] + lnb[r];
}

// --------------------------------- outputs ---------------------------------

__global__ void copy_kernel(const float* __restrict__ src, float* __restrict__ dst, int n) {
  int i = blockIdx.x * blockDim.x + threadIdx.x;
  if (i < n) dst[i] = src[i];
}

// Dt [L x MPA] -> next_x_in [B,C,M,48]
__global__ void writeout_next_kernel(const float* __restrict__ Dt, float* __restrict__ out) {
  size_t i = (size_t)blockIdx.x * blockDim.x + threadIdx.x;
  size_t total = (size_t)B_ * C_ * M_ * 48;
  if (i >= total) return;
  int t = (int)(i % 48);
  int m = (int)((i / 48) % M_);
  int bc = (int)(i / (48 * M_));
  size_t l = (size_t)bc * 48 + t;
  out[i] = Dt[l * MPA_ + m];
}

// --------------------------------- driver ----------------------------------

static inline unsigned cdiv(size_t a, unsigned b) { return (unsigned)((a + b - 1) / b); }

static void launch_gemm(const __bf16* A, const __bf16* Bt, const __bf16* X,
                        float* outF, __bf16* outB, int Mp, int Kp, int L, hipStream_t s) {
  int mb = Mp / 32, lb = L / 64;          // 2x4 tile blocks per wave
  gemm_bf16_wmma<<<dim3((unsigned)(mb * lb)), dim3(32), 0, s>>>(A, Bt, X, outF, outB, Mp, Kp, L, mb);
}

extern "C" void kernel_launch(void* const* d_in, const int* in_sizes, int n_in,
                              void* d_out, int out_size, void* d_ws, size_t ws_size,
                              hipStream_t stream) {
  (void)in_sizes; (void)n_in; (void)out_size; (void)ws_size;
  auto F = [&](int i) { return (const float*)d_in[i]; };

  // input order = setup_inputs() insertion order
  const float* x0     = F(0);
  const float* emb1   = F(1);
  const float* emb2   = F(2);
  const float* lin1_w = F(3);
  const float* lin1_b = F(4);
  const float* lin2_w = F(5);
  const float* lin2_b = F(6);
  const float* assign = F(7);
  // layer param bases: layer0 -> 8, layer1 -> 30
  // [base+0..3]=filt_w(k2,3,7,13) [base+4..7]=filt_b [base+8..11]=gate_w [base+12..15]=gate_b
  // then g1_w,g1_b,g2_w,g2_b,ln_w,ln_b

  float* out = (float*)d_out;
  float* OUT_X   = out;                    // [16,32,1000,48]
  float* OUT_ADP = out + 24576000;         // [1000,1000]
  float* OUT_ASS = out + 25576000;         // [1000,200]
  float* OUT_NXT = out + 25776000;         // [16,32,200,48]

  // workspace carve (deterministic)
  char* wsb = (char*)d_ws;
  size_t off = 0;
  auto carve = [&](size_t bytes) -> void* {
    void* p = wsb + off;
    off += (bytes + 255) & ~(size_t)255;
    return p;
  };
  float*  bufA  = (float*) carve((size_t)B_ * C_ * N_ * 60 * 4);   // layer0 activations
  float*  bufB  = (float*) carve((size_t)B_ * C_ * N_ * 60 * 4);   // layer1 activations
  __bf16* Hb0   = (__bf16*)carve((size_t)30720 * NP_ * 2);
  __bf16* Hb1   = (__bf16*)carve((size_t)30720 * NP_ * 2);
  float*  Dt    = (float*) carve((size_t)30720 * NP_ * 4);
  __bf16* Abf1  = (__bf16*)carve((size_t)NP_ * NP_ * 2);
  __bf16* Abf2  = (__bf16*)carve((size_t)NP_ * NP_ * 2);
  __bf16* AssTb = (__bf16*)carve((size_t)MPA_ * NP_ * 2);
  float*  nv1   = (float*) carve((size_t)N_ * ND_ * 4);
  float*  nv2   = (float*) carve((size_t)N_ * ND_ * 4);
  float*  rs1   = (float*) carve((size_t)N_ * 4);
  float*  rs2   = (float*) carve((size_t)N_ * 4);
  float*  stats = (float*) carve(256);

  // ---- adjacency ----
  nv_kernel<<<cdiv(N_ * ND_, 256), 256, 0, stream>>>(emb1, lin1_w, lin1_b, nv1);
  nv_kernel<<<cdiv(N_ * ND_, 256), 256, 0, stream>>>(emb2, lin2_w, lin2_b, nv2);
  adj_kernel<<<cdiv((size_t)N_ * N_, 256), 256, 0, stream>>>(nv1, nv2, OUT_ADP);
  topk_kernel<<<N_, 128, 0, stream>>>(OUT_ADP);
  rowsum_kernel<<<N_, 128, 0, stream>>>(OUT_ADP, rs1, rs2);
  adjpack_kernel<<<cdiv((size_t)NP_ * NP_, 256), 256, 0, stream>>>(OUT_ADP, rs1, rs2, Abf1, Abf2);
  copy_kernel<<<cdiv(N_ * M_, 256), 256, 0, stream>>>(assign, OUT_ASS, N_ * M_);

  // ---- two encoder layers ----
  for (int i = 0; i < 2; ++i) {
    const float* xin = (i == 0) ? x0 : bufA;
    float* xg        = (i == 0) ? bufA : bufB;
    int Tin  = (i == 0) ? T_ : 60;
    int tlen = (i == 0) ? 60 : 48;
    int L    = B_ * C_ * tlen;              // 30720 / 24576, multiples of 64
    int base = (i == 0) ? 8 : 30;
    const float* g1w = F(base + 16), * g1b = F(base + 17);
    const float* g2w = F(base + 18), * g2b = F(base + 19);

    unsigned gElem = cdiv((size_t)B_ * N_ * tlen, 256);
    inception_kernel<<<gElem, 256, 0, stream>>>(
        xin, Tin, tlen,
        F(base + 0), F(base + 1), F(base + 2), F(base + 3),
        F(base + 4), F(base + 5), F(base + 6), F(base + 7),
        F(base + 8), F(base + 9), F(base + 10), F(base + 11),
        F(base + 12), F(base + 13), F(base + 14), F(base + 15), xg);

    // pack hop0 (gated output) before acc_init overwrites xg in place
    pack_h_kernel<<<cdiv((size_t)L * NP_, 256), 256, 0, stream>>>(xg, Hb0, tlen, L);
    acc_init_kernel<<<gElem, 256, 0, stream>>>(xg, xin, Tin, tlen, g1w, g1b, g2w, g2b);

    for (int m = 0; m < 2; ++m) {
      const __bf16* Am = m ? Abf2 : Abf1;
      const float*  Wm = m ? g2w : g1w;
      // hop1: D = A*h0 ; h1 = 0.05*x + 0.95*D -> f32 Dt + bf16 Hb1
      launch_gemm(Am, Hb0, Hb0, Dt, Hb1, NP_, NP_, L, stream);
      acc_hop_kernel<<<gElem, 256, 0, stream>>>(xg, Dt, Wm, C_, tlen);
      // hop2: D = A*h1 ; h2 = 0.05*x + 0.95*D -> f32 Dt only
      launch_gemm(Am, Hb1, Hb0, Dt, nullptr, NP_, NP_, L, stream);
      acc_hop_kernel<<<gElem, 256, 0, stream>>>(xg, Dt, Wm, 2 * C_, tlen);
    }
  }

  // ---- layernorm (final layer only; earlier x_out values are overwritten) ----
  int cnt = C_ * N_ * 48;                   // 1,536,000 per batch
  stats_zero_kernel<<<1, 32, 0, stream>>>(stats);
  ln_reduce_kernel<<<dim3(256, B_), 256, 0, stream>>>(bufB, stats, cnt);
  ln_apply_kernel<<<cdiv((size_t)B_ * cnt, 256), 256, 0, stream>>>(
      bufB, stats, F(50), F(51), OUT_X, cnt);

  // ---- next_x_in = einsum('mn,bcnt->bcmt', ass^T, x_out) via WMMA ----
  int L2 = B_ * C_ * 48;                    // 24576, multiple of 64
  pack_h_kernel<<<cdiv((size_t)L2 * NP_, 256), 256, 0, stream>>>(OUT_X, Hb0, 48, L2);
  pack_assT_kernel<<<cdiv(MPA_ * NP_, 256), 256, 0, stream>>>(assign, AssTb);
  launch_gemm(AssTb, Hb0, nullptr, Dt, nullptr, MPA_, NP_, L2, stream);
  writeout_next_kernel<<<cdiv((size_t)B_ * C_ * M_ * 48, 256), 256, 0, stream>>>(Dt, OUT_NXT);
}
